// GAU_35158602285680
// MI455X (gfx1250) — matrix operand order repaired and checked by reading, not compiled, not run on hardware
//
#include <hip/hip_runtime.h>
#include <hip/hip_bf16.h>

typedef _Float16 f16;
typedef __attribute__((ext_vector_type(16))) _Float16 v16h;
typedef __attribute__((ext_vector_type(8)))  _Float16 v8h;
typedef __attribute__((ext_vector_type(8)))  float    v8f;

// Problem constants
constexpr int CB   = 20;     // batch
constexpr int CS   = 500;    // seq
constexpr int CD   = 300;    // dim
constexpr int CH   = 600;    // hid
constexpr int CQK  = 128;
constexpr int CROT = 32;
constexpr int CM   = CB * CS;        // 10000 rows (mult of 16)
constexpr int THIST = 128;

// Padded layout constants (K strides mult of 32 halves, B-row counts mult of 64)
constexpr int CKP  = 320;    // padded CD (K of proj GEMMs)
constexpr int CSP  = 512;    // padded CS (attention M/N/K)
constexpr int CHP  = 608;    // padded CH (K of out GEMM / a2 lda)
constexpr int NH2P = 1216;   // padded 2*CH rows of WhT
constexpr int NVP  = 640;    // padded CH rows of vT
constexpr int NWOP = 320;    // padded CD rows of WoutT

// ---------------------------------------------------------------------------
// Block-cooperative GEMM core. A block (8 waves) owns a 128(M) x 64(N) tile:
// the 64xK B strip (packed transposed, ldb==K, so it is contiguous in global)
// is staged once into LDS (row stride K+8 halves to stagger banks), then each
// wave computes a 16x64 strip with 4 accumulators sharing one A fragment.
// Inner loop: 2 global_load_b128 (A) + 8 ds_load_b128 (B) + 4 WMMA per K-step.
// ---------------------------------------------------------------------------
template<int K>
__device__ __forceinline__ void load_btile(f16* __restrict__ Bs,
                                           const f16* __restrict__ src)
{
    constexpr int KL = K + 8;
    constexpr int CPR = K / 8;                 // 8-half chunks per row
    for (int c = threadIdx.x; c < 64 * CPR; c += 256) {
        const int row = c / CPR, off = (c % CPR) * 8;
        *(v8h*)(Bs + row * KL + off) = *(const v8h*)(src + (size_t)row * K + off);
    }
}

template<int K>
__device__ __forceinline__ void wmma_strip_lds(const f16* __restrict__ arow,
                                               const f16* __restrict__ Bs,
                                               v8f acc[4])
{
    constexpr int KL = K + 8;
    const int lane = threadIdx.x & 31;
    const int lh   = lane >> 4;
    const f16* brow0 = Bs + (size_t)(lane & 15) * KL;
#pragma unroll 2
    for (int kb = 0; kb < K; kb += 32) {
        const v8h a0 = *(const v8h*)(arow + kb + lh*8);
        const v8h a1 = *(const v8h*)(arow + kb + lh*8 + 16);
        const v16h af = __builtin_shufflevector(a0, a1,
            0,1,2,3,4,5,6,7,8,9,10,11,12,13,14,15);
#pragma unroll
        for (int t = 0; t < 4; ++t) {
            const f16* brow = brow0 + (size_t)(t*16) * KL;
            const v8h b0 = *(const v8h*)(brow + kb + lh*8);
            const v8h b1 = *(const v8h*)(brow + kb + lh*8 + 16);
            const v16h bf = __builtin_shufflevector(b0, b1,
                0,1,2,3,4,5,6,7,8,9,10,11,12,13,14,15);
            acc[t] = __builtin_amdgcn_wmma_f32_16x16x32_f16(
                false, af, false, bf, (short)0, acc[t], false, false);
        }
    }
}

__device__ __forceinline__ float siluf(float x) { return x / (1.f + expf(-x)); }

// ---------------------------------------------------------------------------
// 1) LayerNorm rows (one block per row)
// ---------------------------------------------------------------------------
__global__ void k_ln(const float* __restrict__ x, const float* __restrict__ g,
                     const float* __restrict__ b, float* __restrict__ ln)
{
    const int row = blockIdx.x;               // 0..CM-1
    __shared__ float ssum[256], ssq[256];
    const float* xr = x + (long)row * CD;
    float s = 0.f, sq = 0.f;
    for (int d = threadIdx.x; d < CD; d += 256) { float v = xr[d]; s += v; sq += v*v; }
    ssum[threadIdx.x] = s; ssq[threadIdx.x] = sq;
    __syncthreads();
    for (int st = 128; st > 0; st >>= 1) {
        if (threadIdx.x < st) { ssum[threadIdx.x] += ssum[threadIdx.x+st];
                                ssq[threadIdx.x]  += ssq[threadIdx.x+st]; }
        __syncthreads();
    }
    const float mean = ssum[0] / (float)CD;
    const float var  = ssq[0] / (float)CD - mean*mean;
    const float rstd = rsqrtf(var + 1e-5f);
    for (int d = threadIdx.x; d < CD; d += 256)
        ln[(long)row*CD + d] = (xr[d] - mean) * rstd * g[d] + b[d];
}

// ---------------------------------------------------------------------------
// 2) Time-shift first half, pack to f16, K padded to CKP (zeros)
// ---------------------------------------------------------------------------
__global__ void k_shift_pack(const float* __restrict__ ln, f16* __restrict__ nx)
{
    const long idx = (long)blockIdx.x * blockDim.x + threadIdx.x;
    if (idx >= (long)CM * CKP) return;
    const int d = (int)(idx % CKP);
    const long row = idx / CKP;
    const int s = (int)(row % CS);
    float v = 0.f;
    if (d < CD) {
        if (d < CD/2) v = (s == 0) ? 0.f : ln[(row-1)*CD + d];
        else          v = ln[row*CD + d];
    }
    nx[idx] = (f16)v;
}

// ---------------------------------------------------------------------------
// 3) Transposed weight convert: Wt[n*ldt+k] = (k<Kd && n<Nd) ? W[k*Nd+n] : 0
// ---------------------------------------------------------------------------
__global__ void k_cvt_t(const float* __restrict__ W, f16* __restrict__ Wt,
                        int Kd, int Nd, int ldt, int nrows)
{
    const long idx = (long)blockIdx.x * blockDim.x + threadIdx.x;
    if (idx >= (long)nrows * ldt) return;
    const int n = (int)(idx / ldt);
    const int k = (int)(idx % ldt);
    Wt[idx] = (k < Kd && n < Nd) ? (f16)W[(long)k * Nd + n] : (f16)0.f;
}

__global__ void k_zero_h(f16* __restrict__ p, long n)
{
    const long i = (long)blockIdx.x * blockDim.x + threadIdx.x;
    if (i < n) p[i] = (f16)0.f;
}

// ---------------------------------------------------------------------------
// 4) h = silu(nx@Wh + bh); split -> vT (transposed, f16, padded), gate (f32)
//    Block tile: 128(M) x 64(N); grid = 79 Mblocks x 19 Nstrips
// ---------------------------------------------------------------------------
__global__ void k_gemm_h(const f16* __restrict__ nx, const f16* __restrict__ WhT,
                         const float* __restrict__ bh,
                         f16* __restrict__ vT, float* __restrict__ gate)
{
    constexpr int K = CKP, NS = 19, MT = CM/16;        // 320, 19, 625
    __shared__ __align__(16) f16 Bs[64 * (K + 8)];
    const int ns = blockIdx.x % NS;
    const int mb = blockIdx.x / NS;
    load_btile<K>(Bs, WhT + (size_t)ns * 64 * K);
    __syncthreads();
    int mt = mb*8 + (threadIdx.x >> 5);
    const bool active = mt < MT;
    mt = min(mt, MT-1);
    const int lane = threadIdx.x & 31;
    const int lh = lane >> 4;
    const f16* arow = nx + (size_t)(mt*16 + (lane & 15)) * K;
    v8f acc[4] = {};
    wmma_strip_lds<K>(arow, Bs, acc);
    if (!active) return;
#pragma unroll
    for (int t = 0; t < 4; ++t) {
        const int col = ns*64 + t*16 + (lane & 15);
#pragma unroll
        for (int r = 0; r < 8; ++r) {
            const int row = mt*16 + r + lh*8;
            if (col < 2*CH) {
                const float h = siluf(acc[t][r] + bh[col]);
                const int b = row / CS, s = row % CS;
                if (col < CH) vT[((size_t)b*NVP + col)*CSP + s] = (f16)h;
                else          gate[(long)row*CH + (col - CH)] = h;
            }
        }
    }
}

// ---------------------------------------------------------------------------
// 5) qk = silu(nx@Wqk + bqk)  (f32)
// ---------------------------------------------------------------------------
__global__ void k_gemm_qk(const f16* __restrict__ nx, const f16* __restrict__ WqkT,
                          const float* __restrict__ bqk, float* __restrict__ qk)
{
    constexpr int K = CKP, NS = CQK/64, MT = CM/16;    // 320, 2, 625
    __shared__ __align__(16) f16 Bs[64 * (K + 8)];
    const int ns = blockIdx.x % NS;
    const int mb = blockIdx.x / NS;
    load_btile<K>(Bs, WqkT + (size_t)ns * 64 * K);
    __syncthreads();
    int mt = mb*8 + (threadIdx.x >> 5);
    const bool active = mt < MT;
    mt = min(mt, MT-1);
    const int lane = threadIdx.x & 31;
    const int lh = lane >> 4;
    const f16* arow = nx + (size_t)(mt*16 + (lane & 15)) * K;
    v8f acc[4] = {};
    wmma_strip_lds<K>(arow, Bs, acc);
    if (!active) return;
#pragma unroll
    for (int t = 0; t < 4; ++t) {
        const int col = ns*64 + t*16 + (lane & 15);
#pragma unroll
        for (int r = 0; r < 8; ++r) {
            const int row = mt*16 + r + lh*8;
            qk[(long)row*CQK + col] = siluf(acc[t][r] + bqk[col]);
        }
    }
}

// ---------------------------------------------------------------------------
// 6) rotary -> q_h, k_h in padded per-batch layout (CSP x CQK), pad rows zero
// ---------------------------------------------------------------------------
__global__ void k_rotary(const float* __restrict__ qk, const float* __restrict__ gamma,
                         const float* __restrict__ beta,
                         f16* __restrict__ qh, f16* __restrict__ kh)
{
    const long idx = (long)blockIdx.x * blockDim.x + threadIdx.x;
    if (idx >= (long)CB * CSP * CQK) return;
    const int d = (int)(idx % CQK);
    const int s = (int)((idx / CQK) % CSP);
    const int b = (int)(idx / ((long)CSP * CQK));
    if (s >= CS) { qh[idx] = (f16)0.f; kh[idx] = (f16)0.f; return; }
    const long src = ((long)b*CS + s) * CQK;
    const float qv = qk[src + d];
    float qo = qv * gamma[d]       + beta[d];
    float ko = qv * gamma[CQK + d] + beta[CQK + d];
    if (d < CROT) {
        const int part = d ^ 1;
        const float pv = qk[src + part];
        const float qp = pv * gamma[part]       + beta[part];
        const float kp = pv * gamma[CQK + part] + beta[CQK + part];
        const float inv = powf(10000.f, -(float)(d & ~1) / (float)CROT);
        const float ang = (float)s * inv;
        const float c = cosf(ang), sn = sinf(ang);
        const float qrot = (d & 1) ? qp : -qp;
        const float krot = (d & 1) ? kp : -kp;
        qo = qo * c + qrot * sn;
        ko = ko * c + krot * sn;
    }
    qh[idx] = (f16)qo;
    kh[idx] = (f16)ko;
}

// ---------------------------------------------------------------------------
// 7) T5 relative position bias (S x S)
// ---------------------------------------------------------------------------
__global__ void k_t5bias(const float* __restrict__ emb, float* __restrict__ bias)
{
    const int idx = blockIdx.x * blockDim.x + threadIdx.x;
    if (idx >= CS*CS) return;
    const int i = idx / CS, j = idx % CS;
    const int n = i - j;
    const int ret = (n < 0) ? 16 : 0;
    const int na = abs(n);
    int bucket;
    if (na < 8) bucket = ret + na;
    else {
        int vil = 8 + (int)(logf((float)na / 8.f) / logf(16.f) * 8.f);
        vil = min(vil, 15);
        bucket = ret + vil;
    }
    bias[idx] = emb[bucket] * 11.313708499f;   // sqrt(128)
}

// ---------------------------------------------------------------------------
// 8) attn = relu((q@k^T + bias)/S)^2 * mask2 -> f16 (CSP x CSP, pad = 0)
//    grid = CB x 8 Nstrips x 4 Mblocks (exact fit, no idle waves)
// ---------------------------------------------------------------------------
__global__ void k_gemm_sim(const f16* __restrict__ qh, const f16* __restrict__ kh,
                           const float* __restrict__ bias, const float* __restrict__ mask2,
                           f16* __restrict__ attn)
{
    constexpr int K = CQK, NS = CSP/64, MBLK = CSP/128;  // 128, 8, 4
    __shared__ __align__(16) f16 Bs[64 * (K + 8)];
    const int mb = blockIdx.x % MBLK;
    const int ns = (blockIdx.x / MBLK) % NS;
    const int b  = blockIdx.x / (MBLK * NS);
    load_btile<K>(Bs, kh + (size_t)b*CSP*CQK + (size_t)ns * 64 * K);
    __syncthreads();
    const int mt = mb*8 + (threadIdx.x >> 5);
    const int lane = threadIdx.x & 31;
    const int lh = lane >> 4;
    const f16* arow = qh + (size_t)b*CSP*CQK + (size_t)(mt*16 + (lane & 15)) * K;
    v8f acc[4] = {};
    wmma_strip_lds<K>(arow, Bs, acc);
    f16* ab = attn + (size_t)b * CSP * CSP;
#pragma unroll
    for (int t = 0; t < 4; ++t) {
        const int col = ns*64 + t*16 + (lane & 15);
#pragma unroll
        for (int r = 0; r < 8; ++r) {
            const int row = mt*16 + r + lh*8;
            float a = 0.f;
            if (row < CS && col < CS) {
                const float v = acc[t][r] + bias[row*CS + col];
                a = fmaxf(v * (1.f/(float)CS), 0.f);
                a = a * a * mask2[row*CS + col];
            }
            ab[(size_t)row*CSP + col] = (f16)a;        // fills padding with 0
        }
    }
}

// ---------------------------------------------------------------------------
// 9) oa = attn @ v  (B from packed vT, K=CSP); grid = CB x 10 x 4
// ---------------------------------------------------------------------------
__global__ void k_gemm_av(const f16* __restrict__ attn, const f16* __restrict__ vT,
                          float* __restrict__ oa)
{
    constexpr int K = CSP, NS = NVP/64, MBLK = CSP/128;  // 512, 10, 4
    __shared__ __align__(16) f16 Bs[64 * (K + 8)];
    const int mb = blockIdx.x % MBLK;
    const int ns = (blockIdx.x / MBLK) % NS;
    const int b  = blockIdx.x / (MBLK * NS);
    load_btile<K>(Bs, vT + (size_t)b*NVP*CSP + (size_t)ns * 64 * K);
    __syncthreads();
    const int mt = mb*8 + (threadIdx.x >> 5);
    const int lane = threadIdx.x & 31;
    const int lh = lane >> 4;
    const f16* arow = attn + (size_t)b*CSP*CSP + (size_t)(mt*16 + (lane & 15)) * K;
    v8f acc[4] = {};
    wmma_strip_lds<K>(arow, Bs, acc);
#pragma unroll
    for (int t = 0; t < 4; ++t) {
        const int col = ns*64 + t*16 + (lane & 15);
#pragma unroll
        for (int r = 0; r < 8; ++r) {
            const int row = mt*16 + r + lh*8;
            if (row < CS && col < CH)
                oa[((size_t)b*CS + row)*CH + col] = acc[t][r];
        }
    }
}

// ---------------------------------------------------------------------------
// Gate-mask chain
// ---------------------------------------------------------------------------
__global__ void k_init_hist(int* __restrict__ hist, unsigned* __restrict__ gmax)
{
    const int i = blockIdx.x * blockDim.x + threadIdx.x;
    if (i < CB*(THIST+1)) hist[i] = 0;
    if (i < CB) gmax[i] = 0u;
}

__global__ void k_hist(const float* __restrict__ gate, int* __restrict__ hist,
                       unsigned* __restrict__ gmax)
{
    constexpr int PER_B = CS * CH;                   // 300000
    constexpr int BLKS  = (PER_B + 4095) / 4096;     // 74
    const int b   = blockIdx.x / BLKS;
    const int blk = blockIdx.x % BLKS;
    __shared__ int sh[THIST+1];
    __shared__ unsigned smax;
    for (int i = threadIdx.x; i <= THIST; i += 256) sh[i] = 0;
    if (threadIdx.x == 0) smax = 0u;
    __syncthreads();
    const float* gb = gate + (long)b * PER_B;
    const int base = blk * 4096;
    const int end  = min(base + 4096, PER_B);
    unsigned lmax = 0u;
    for (int i = base + threadIdx.x; i < end; i += 256) {
        const float gv = gb[i];
        const float ag = fabsf(gv);
        int bin = (int)floorf(ag);
        bin = min(max(bin, 0), THIST);
        atomicAdd(&sh[bin], 1);
        unsigned bits = __float_as_uint(gv);
        unsigned enc = (bits & 0x80000000u) ? ~bits : (bits | 0x80000000u);
        lmax = max(lmax, enc);
    }
    atomicMax(&smax, lmax);
    __syncthreads();
    for (int i = threadIdx.x; i <= THIST; i += 256)
        if (sh[i]) atomicAdd(&hist[b*(THIST+1) + i], sh[i]);
    if (threadIdx.x == 0) atomicMax(&gmax[b], smax);
}

__global__ void k_trim(const int* __restrict__ hist, const unsigned* __restrict__ gmaxEnc,
                       float* __restrict__ trim)
{
    const int b = threadIdx.x;
    if (b >= CB) return;
    const unsigned enc = gmaxEnc[b];
    const unsigned bits = (enc & 0x80000000u) ? (enc ^ 0x80000000u) : ~enc;
    const float gmax = floorf(__uint_as_float(bits));
    int cum = 0, best = 0;
    for (int t = THIST; t >= 1; --t) {
        cum += hist[b*(THIST+1) + t];
        if (best == 0 && cum > 90000 && (float)t <= gmax) best = t;
    }
    trim[b] = (float)max(best, 1);
}

__global__ void k_counts(const float* __restrict__ gate, const float* __restrict__ trim,
                         int* __restrict__ counts)
{
    const int idx = blockIdx.x * blockDim.x + threadIdx.x;
    if (idx >= CB * 100 * 120) return;
    const int b = idx / 12000;
    const int rem = idx % 12000;
    const int p = rem / 120, qq = rem % 120;
    const float tr = trim[b];
    int cnt = 0;
#pragma unroll
    for (int r = 0; r < 4; ++r)
#pragma unroll
        for (int c = 0; c < 4; ++c) {
            const float ag = fabsf(gate[((long)b*CS + (p*5 + r))*CH + (qq*5 + c)]);
            cnt += (ag >= tr) ? 1 : 0;
        }
    counts[idx] = cnt;
}

__global__ void k_t2(const int* __restrict__ counts, int* __restrict__ t2g)
{
    const int b = blockIdx.x;
    __shared__ int sh[17];
    __shared__ int smaxv;
    if (threadIdx.x < 17) sh[threadIdx.x] = 0;
    if (threadIdx.x == 0) smaxv = 0;
    __syncthreads();
    const int* cb = counts + b * 12000;
    for (int i = threadIdx.x; i < 12000; i += 256) {
        const int v = cb[i];
        atomicAdd(&sh[v], 1);
        atomicMax(&smaxv, v);
    }
    __syncthreads();
    if (threadIdx.x == 0) {
        const int cmax = smaxv;
        int cum = 0, t2 = 0;
        bool found = false;
        for (int t = 16; t >= 1; --t) {
            cum += sh[t];
            if (!found && t <= cmax && cum > 3600) { t2 = t; found = true; }
        }
        t2g[b] = found ? t2 : cmax;
    }
}

// a2 covers padded CHP columns (zeros in pad) so out-GEMM needs no K guard
__global__ void k_apply(const float* __restrict__ oa, const float* __restrict__ gate,
                        const int* __restrict__ counts, const int* __restrict__ t2,
                        f16* __restrict__ a2)
{
    const long idx = (long)blockIdx.x * blockDim.x + threadIdx.x;
    if (idx >= (long)CM * CHP) return;
    const int j = (int)(idx % CHP);
    const long row = idx / CHP;
    if (j >= CH) { a2[idx] = (f16)0.f; return; }
    const int b = (int)(row / CS);
    const int s = (int)(row % CS);
    const int r = s % 5, c = j % 5;
    float m = 0.f;
    if (r < 4 && c < 4) {
        const int p = s / 5, qq = j / 5;
        m = (counts[b*12000 + p*120 + qq] >= t2[b]) ? 1.f : 0.25f;
    }
    a2[idx] = (f16)(m * oa[row*CH + j] * gate[row*CH + j]);
}

// ---------------------------------------------------------------------------
// 10) out = a2 @ Wout + bout + x   (grid = 79 Mblocks x 5 Nstrips)
// ---------------------------------------------------------------------------
__global__ void k_gemm_out(const f16* __restrict__ a2, const f16* __restrict__ WoutT,
                           const float* __restrict__ bout, const float* __restrict__ x,
                           float* __restrict__ out)
{
    constexpr int K = CHP, NS = NWOP/64, MT = CM/16;   // 608, 5, 625
    __shared__ __align__(16) f16 Bs[64 * (K + 8)];
    const int ns = blockIdx.x % NS;
    const int mb = blockIdx.x / NS;
    load_btile<K>(Bs, WoutT + (size_t)ns * 64 * K);
    __syncthreads();
    int mt = mb*8 + (threadIdx.x >> 5);
    const bool active = mt < MT;
    mt = min(mt, MT-1);
    const int lane = threadIdx.x & 31;
    const int lh = lane >> 4;
    const f16* arow = a2 + (size_t)(mt*16 + (lane & 15)) * K;
    v8f acc[4] = {};
    wmma_strip_lds<K>(arow, Bs, acc);
    if (!active) return;
#pragma unroll
    for (int t = 0; t < 4; ++t) {
        const int col = ns*64 + t*16 + (lane & 15);
#pragma unroll
        for (int r = 0; r < 8; ++r) {
            const int row = mt*16 + r + lh*8;
            if (col < CD)
                out[(long)row*CD + col] = acc[t][r] + bout[col] + x[(long)row*CD + col];
        }
    }
}

// ---------------------------------------------------------------------------
extern "C" void kernel_launch(void* const* d_in, const int* in_sizes, int n_in,
                              void* d_out, int out_size, void* d_ws, size_t ws_size,
                              hipStream_t stream)
{
    const float* x      = (const float*)d_in[0];
    const float* mask2  = (const float*)d_in[1];
    const float* ln_g   = (const float*)d_in[2];
    const float* ln_b   = (const float*)d_in[3];
    const float* Wh     = (const float*)d_in[4];
    const float* bh     = (const float*)d_in[5];
    const float* Wqk    = (const float*)d_in[6];
    const float* bqk    = (const float*)d_in[7];
    const float* gamma  = (const float*)d_in[8];
    const float* beta   = (const float*)d_in[9];
    const float* relemb = (const float*)d_in[10];
    const float* Wout   = (const float*)d_in[11];
    const float* bout   = (const float*)d_in[12];
    float* out = (float*)d_out;

    char* ws = (char*)d_ws;
    size_t off = 0;
    auto take = [&](size_t bytes) -> char* {
        size_t o = (off + 255) & ~(size_t)255;
        off = o + bytes;
        return ws + o;
    };

    float*    ln_buf  = (float*)   take((size_t)CM*CD*4);
    f16*      nx_h    = (f16*)     take((size_t)CM*CKP*2);
    f16*      WhT_h   = (f16*)     take((size_t)NH2P*CKP*2);
    f16*      WqkT_h  = (f16*)     take((size_t)CQK*CKP*2);
    f16*      WoutT_h = (f16*)     take((size_t)NWOP*CHP*2);
    f16*      vT_h    = (f16*)     take((size_t)CB*NVP*CSP*2);
    float*    gate_f  = (float*)   take((size_t)CM*CH*4);
    float*    qk_f    = (float*)   take((size_t)CM*CQK*4);
    f16*      q_h     = (f16*)     take((size_t)CB*CSP*CQK*2);
    f16*      k_h     = (f16*)     take((size_t)CB*CSP*CQK*2);
    float*    bias_f  = (float*)   take((size_t)CS*CS*4);
    f16*      attn_h  = (f16*)     take((size_t)CB*CSP*CSP*2);
    float*    oa_f    = (float*)   take((size_t)CM*CH*4);
    f16*      a2_h    = (f16*)     take((size_t)CM*CHP*2);
    int*      hist    = (int*)     take((size_t)CB*(THIST+1)*4);
    unsigned* gmax    = (unsigned*)take((size_t)CB*4);
    float*    trim    = (float*)   take((size_t)CB*4);
    int*      counts  = (int*)     take((size_t)CB*100*120*4);
    int*      t2      = (int*)     take((size_t)CB*4);
    (void)ws_size; (void)in_sizes; (void)n_in; (void)out_size;

    auto cdiv = [](long a, long b) { return (int)((a + b - 1) / b); };

    // 1) layernorm
    k_ln<<<CM, 256, 0, stream>>>(x, ln_g, ln_b, ln_buf);
    // 2) shift + pack f16 (padded K)
    k_shift_pack<<<cdiv((long)CM*CKP, 256), 256, 0, stream>>>(ln_buf, nx_h);
    // 3) transposed weight converts (zero-padded)
    k_cvt_t<<<cdiv((long)NH2P*CKP, 256), 256, 0, stream>>>(Wh,   WhT_h,   CD, 2*CH, CKP, NH2P);
    k_cvt_t<<<cdiv((long)CQK*CKP,  256), 256, 0, stream>>>(Wqk,  WqkT_h,  CD, CQK,  CKP, CQK);
    k_cvt_t<<<cdiv((long)NWOP*CHP, 256), 256, 0, stream>>>(Wout, WoutT_h, CH, CD,   CHP, NWOP);
    // vT padding must be zero (gemm_h only writes valid (c,s))
    k_zero_h<<<cdiv((long)CB*NVP*CSP, 256), 256, 0, stream>>>(vT_h, (long)CB*NVP*CSP);
    // 4) h GEMM (M=10000, N=1200, K=320): 79 x 19 blocks
    k_gemm_h<<<79*19, 256, 0, stream>>>(nx_h, WhT_h, bh, vT_h, gate_f);
    // 5) qk GEMM (M=10000, N=128, K=320): 79 x 2 blocks
    k_gemm_qk<<<79*2, 256, 0, stream>>>(nx_h, WqkT_h, bqk, qk_f);
    // 6) rotary (writes padded rows as zero)
    k_rotary<<<cdiv((long)CB*CSP*CQK, 256), 256, 0, stream>>>(qk_f, gamma, beta, q_h, k_h);
    // 7) T5 bias
    k_t5bias<<<cdiv(CS*CS, 256), 256, 0, stream>>>(relemb, bias_f);
    // 8) attention scores (per-batch 512x512, K=128): 20 x 8 x 4 blocks
    k_gemm_sim<<<CB*8*4, 256, 0, stream>>>(q_h, k_h, bias_f, mask2, attn_h);
    // 9) attn @ v (per-batch M=512, K=512): 20 x 10 x 4 blocks
    k_gemm_av<<<CB*10*4, 256, 0, stream>>>(attn_h, vT_h, oa_f);
    // gate-mask chain
    k_init_hist<<<cdiv(CB*(THIST+1), 256), 256, 0, stream>>>(hist, gmax);
    {
        constexpr int BLKS = (CS*CH + 4095)/4096;     // 74 per batch
        k_hist<<<CB * BLKS, 256, 0, stream>>>(gate_f, hist, gmax);
    }
    k_trim<<<1, 32, 0, stream>>>(hist, gmax, trim);
    k_counts<<<cdiv(CB*100*120, 256), 256, 0, stream>>>(gate_f, trim, counts);
    k_t2<<<CB, 256, 0, stream>>>(counts, t2);
    k_apply<<<cdiv((long)CM*CHP, 256), 256, 0, stream>>>(oa_f, gate_f, counts, t2, a2_h);
    // 10) output GEMM + residual (M=10000, N=300, K=608): 79 x 5 blocks
    k_gemm_out<<<79*5, 256, 0, stream>>>(a2_h, WoutT_h, bout, x, out);
}
